// CrossResolutionAttention_80985903333598
// MI455X (gfx1250) — matrix operand order repaired and checked by reading, not compiled, not run on hardware
//
#include <hip/hip_runtime.h>
#include <hip/hip_bf16.h>
#include <stdint.h>

// ---------------------------------------------------------------------------
// CDNA5 / gfx1250 fused cross-resolution attention
//   B=8, NB=3, L=4096, D=256, H=4, hd=64
// ---------------------------------------------------------------------------

typedef __attribute__((ext_vector_type(16))) __bf16 v16bf;
typedef __attribute__((ext_vector_type(8)))  float  v8f;

union FragBF { v16bf v; uint4 u[2]; };

// Async global->LDS path (CDNA5): guarded so either toolchain compiles.
#if defined(__has_builtin)
#if __has_builtin(__builtin_amdgcn_global_load_async_to_lds_b128) && \
    __has_builtin(__builtin_amdgcn_s_wait_asynccnt)
#define USE_ASYNC_LDS 1
#endif
#endif
#ifndef USE_ASYNC_LDS
#define USE_ASYNC_LDS 0
#endif

#if USE_ASYNC_LDS
// Builtin expects pointers to 16-byte int vectors in AS1 (global) / AS3 (LDS).
typedef int v4i_vs __attribute__((__vector_size__(16)));
typedef __attribute__((address_space(1))) v4i_vs* as1_v4i;
typedef __attribute__((address_space(3))) v4i_vs* as3_v4i;
#endif

__device__ __forceinline__ unsigned short f2bf(float f) {
  unsigned u = __float_as_uint(f);
  u += 0x7FFFu + ((u >> 16) & 1u);     // round-to-nearest-even
  return (unsigned short)(u >> 16);
}
__device__ __forceinline__ float bf2f(unsigned short h) {
  return __uint_as_float(((unsigned)h) << 16);
}

#define D_DIM  256
#define LFULL  4096
#define NBR    3
#define SITES  16
#define MROWS  48           // SITES * NBR
#define ASTR   264          // 256 + 8 bf16 pad (bank rotation)
#define WSTR   264

// ---------------------------------------------------------------------------
// Kernel 1: pack W^T to bf16:  Wt[mat][n][k] = W[k][n]
// ---------------------------------------------------------------------------
__global__ __launch_bounds__(256) void prep_weights(
    const float* __restrict__ Wq, const float* __restrict__ Wk,
    const float* __restrict__ Wv, const float* __restrict__ Wo,
    unsigned short* __restrict__ Wt)
{
  int idx = blockIdx.x * 256 + threadIdx.x;      // 0 .. 4*65536-1
  int mat = idx >> 16;
  int n   = (idx >> 8) & 255;
  int k   = idx & 255;
  const float* W = (mat == 0) ? Wq : (mat == 1) ? Wk : (mat == 2) ? Wv : Wo;
  Wt[idx] = f2bf(W[k * 256 + n]);
}

// ---------------------------------------------------------------------------
// Cooperative W^T (256x256 bf16) -> LDS.  g-linear mapping: per wave-
// instruction, 32 lanes cover 512 contiguous bytes of global memory and one
// contiguous LDS row segment.  Async variant bypasses VGPRs (ASYNCcnt).
// ---------------------------------------------------------------------------
__device__ __forceinline__ void issue_w_load(
    const unsigned short* __restrict__ wsrc, unsigned short* Wl, int tid)
{
#pragma unroll 4
  for (int i = 0; i < 32; ++i) {
    const int g = i * 256 + tid;          // uint4 index, 0..8191
    const int n = g >> 5;                 // LDS row (32 uint4 per 256-bf16 row)
    const int c = g & 31;
#if USE_ASYNC_LDS
    __builtin_amdgcn_global_load_async_to_lds_b128(
        (as1_v4i)(wsrc + (size_t)g * 8),
        (as3_v4i)(Wl + n * WSTR + c * 8),
        0, 0);
#else
    *(uint4*)(Wl + n * WSTR + c * 8) = *(const uint4*)(wsrc + (size_t)g * 8);
#endif
  }
}

__device__ __forceinline__ void wait_w_load() {
#if USE_ASYNC_LDS
  __builtin_amdgcn_s_wait_asynccnt(0);
#endif
}

// ---------------------------------------------------------------------------
// Per-wave WMMA GEMM of the 48x256 LDS A-tile times 256x256 LDS W^T.
// Wave w owns ntiles {2w, 2w+1}; mtiles 0..2.  acc = A @ W (f32).
// ---------------------------------------------------------------------------
__device__ __forceinline__ void gemm48x256(
    const unsigned short* At, const unsigned short* Wl,
    int lane, int wave, v8f acc[3][2])
{
  const int mrow = lane & 15;
  const int hi   = lane >> 4;
  for (int k0 = 0; k0 < 256; k0 += 32) {
    FragBF af[3];
#pragma unroll
    for (int mt = 0; mt < 3; ++mt) {
      // A 16-bit layout: lane m holds K {k0+hi*8 .. +7} and {k0+16+hi*8 .. +7}
      const uint4* ap = (const uint4*)(At + (mt * 16 + mrow) * ASTR + k0 + hi * 8);
      af[mt].u[0] = ap[0];
      af[mt].u[1] = ap[2];                      // +16 bf16 = +32B
    }
#pragma unroll
    for (int nt = 0; nt < 2; ++nt) {
      // B 16-bit layout: lane n holds K {k0+hi*8 .. +7} and {k0+16+hi*8 .. +7}
      const uint4* bp = (const uint4*)(Wl + ((wave * 2 + nt) * 16 + mrow) * WSTR + k0 + hi * 8);
      FragBF bfr; bfr.u[0] = bp[0]; bfr.u[1] = bp[2];
#pragma unroll
      for (int mt = 0; mt < 3; ++mt) {
        acc[mt][nt] = __builtin_amdgcn_wmma_f32_16x16x32_bf16(
            false, af[mt].v, false, bfr.v, (short)0, acc[mt][nt], false, false);
      }
    }
  }
}

// ---------------------------------------------------------------------------
// Kernel 2: fused interp + QKV GEMMs + 3x3 branch attention + O GEMM
// grid = B * (L/SITES) = 8 * 256 = 2048 blocks of 256 threads (8 waves)
// ---------------------------------------------------------------------------
__global__ __launch_bounds__(256) void fused_attn(
    const float* __restrict__ br0, const float* __restrict__ br1,
    const float* __restrict__ br2,
    const float* __restrict__ bq, const float* __restrict__ bk,
    const float* __restrict__ bv, const float* __restrict__ bo,
    const unsigned short* __restrict__ Wt,
    float* __restrict__ y)                       // y[(b*L+l)*3+nb][n] f32
{
  __shared__ __attribute__((aligned(16))) unsigned short At[MROWS * ASTR];      // 25,344 B
  __shared__ __attribute__((aligned(16))) unsigned short Wl[D_DIM * WSTR];      // 135,168 B
  __shared__ __attribute__((aligned(16))) unsigned short QKV[3 * MROWS * ASTR]; // 76,032 B

  const int tid  = threadIdx.x;
  const int lane = tid & 31;
  const int wave = tid >> 5;
  const int bb   = blockIdx.x >> 8;              // batch
  const int l0   = (blockIdx.x & 255) << 4;      // site base

  // Start Wq DMA immediately; it overlaps the entire interpolation phase.
  issue_w_load(Wt, Wl, tid);

  // ---- Phase 0: interpolated A-tile load (f32 -> bf16) --------------------
  for (int r = 0; r < MROWS; ++r) {
    const int s  = r / 3;
    const int nb = r % 3;
    const int l  = l0 + s;
    float val;
    if (nb == 0) {
      val = br0[((size_t)bb * LFULL + l) * D_DIM + tid];
    } else {
      const int   Lin   = (nb == 1) ? 2048 : 1024;
      const float scale = (nb == 1) ? 0.5f : 0.25f;
      float src = ((float)l + 0.5f) * scale - 0.5f;
      src = fminf(fmaxf(src, 0.0f), (float)(Lin - 1));
      int i0 = (int)src;
      int i1 = i0 + 1; if (i1 > Lin - 1) i1 = Lin - 1;
      float w = src - (float)i0;
      const float* sp = (nb == 1) ? br1 : br2;
      size_t base = (size_t)bb * Lin * D_DIM;
      val = sp[base + (size_t)i0 * D_DIM + tid] * (1.0f - w)
          + sp[base + (size_t)i1 * D_DIM + tid] * w;
    }
    At[r * ASTR + tid] = f2bf(val);
  }
  wait_w_load();
  __syncthreads();

  // ---- Phase 1: Q, K, V projections --------------------------------------
  const int mrow = lane & 15;
  const int hi   = lane >> 4;
  for (int p = 0; p < 3; ++p) {
    v8f zero = {};
    v8f acc[3][2];
#pragma unroll
    for (int mt = 0; mt < 3; ++mt) { acc[mt][0] = zero; acc[mt][1] = zero; }

    gemm48x256(At, Wl, lane, wave, acc);
    __syncthreads();                       // all waves done reading Wl(p)

    // Kick off next weight DMA (Wk, Wv, then Wo); overlaps the bias+store
    // below, and for p==2 it also overlaps the whole attention phase.
    issue_w_load(Wt + (size_t)(p + 1) * 65536, Wl, tid);

    const float* bias = (p == 0) ? bq : (p == 1) ? bk : bv;
#pragma unroll
    for (int mt = 0; mt < 3; ++mt) {
#pragma unroll
      for (int nt = 0; nt < 2; ++nt) {
        const int n = (wave * 2 + nt) * 16 + mrow;
        const float bval = bias[n];
#pragma unroll
        for (int j = 0; j < 8; ++j) {
          const int m = mt * 16 + j + 8 * hi;    // C/D layout: vgpr j, lane-half
          QKV[(p * MROWS + m) * ASTR + n] = f2bf(acc[mt][nt][j] + bval);
        }
      }
    }
    if (p < 2) wait_w_load();              // W(p+1) must be resident next iter
    __syncthreads();                       // QKV(p) visible
  }

  // ---- Phase 2: 3x3 attention over branches, per (site, head) ------------
  // (Wo DMA still in flight; attention never touches Wl.)
  if (tid < 64) {
    const int s = tid & 15;
    const int h = tid >> 4;
    const int r0 = s * 3;
    float lg[3][3] = {{0.f,0.f,0.f},{0.f,0.f,0.f},{0.f,0.f,0.f}};
    for (int d = 0; d < 64; ++d) {
      const int off = h * 64 + d;
      float qv[3], kv[3];
#pragma unroll
      for (int n = 0; n < 3; ++n) {
        qv[n] = bf2f(QKV[(0 * MROWS + r0 + n) * ASTR + off]);
        kv[n] = bf2f(QKV[(1 * MROWS + r0 + n) * ASTR + off]);
      }
#pragma unroll
      for (int n = 0; n < 3; ++n)
#pragma unroll
        for (int m = 0; m < 3; ++m) lg[n][m] += qv[n] * kv[m];
    }
    float pm[3][3];
#pragma unroll
    for (int n = 0; n < 3; ++n) {
      float a0 = lg[n][0] * 0.125f, a1 = lg[n][1] * 0.125f, a2 = lg[n][2] * 0.125f;
      float mx = fmaxf(a0, fmaxf(a1, a2));
      pm[n][0] = __expf(a0 - mx); pm[n][1] = __expf(a1 - mx); pm[n][2] = __expf(a2 - mx);
      float inv = 1.0f / (pm[n][0] + pm[n][1] + pm[n][2]);
      pm[n][0] *= inv; pm[n][1] *= inv; pm[n][2] *= inv;
    }
    for (int d = 0; d < 64; ++d) {
      const int off = h * 64 + d;
      float vv[3];
#pragma unroll
      for (int m = 0; m < 3; ++m) vv[m] = bf2f(QKV[(2 * MROWS + r0 + m) * ASTR + off]);
#pragma unroll
      for (int n = 0; n < 3; ++n) {
        float o = pm[n][0] * vv[0] + pm[n][1] * vv[1] + pm[n][2] * vv[2];
        At[(r0 + n) * ASTR + off] = f2bf(o);     // reuse A-tile as attn output
      }
    }
  }
  wait_w_load();                           // Wo resident before O-GEMM
  __syncthreads();

  // ---- Phase 3: output projection, write y -------------------------------
  v8f zero = {};
  v8f acc[3][2];
#pragma unroll
  for (int mt = 0; mt < 3; ++mt) { acc[mt][0] = zero; acc[mt][1] = zero; }

  gemm48x256(At, Wl, lane, wave, acc);

#pragma unroll
  for (int mt = 0; mt < 3; ++mt) {
#pragma unroll
    for (int nt = 0; nt < 2; ++nt) {
      const int n = (wave * 2 + nt) * 16 + mrow;
      const float bval = bo[n];
#pragma unroll
      for (int j = 0; j < 8; ++j) {
        const int m  = mt * 16 + j + 8 * hi;
        const int l  = l0 + m / 3;
        const int nb = m % 3;
        y[(((size_t)bb * LFULL + l) * NBR + nb) * D_DIM + n] = acc[mt][nt][j] + bval;
      }
    }
  }
}

// ---------------------------------------------------------------------------
// Kernel 3: resample y (L=4096) back to branch lengths; concat into d_out
// ---------------------------------------------------------------------------
__global__ __launch_bounds__(256) void resample_out(
    const float* __restrict__ y, float* __restrict__ out)
{
  const long long S0 = 8LL * 4096 * 256;
  const long long S1 = 8LL * 2048 * 256;
  long long idx = (long long)blockIdx.x * 256 + threadIdx.x;

  int nb, outLen; long long j;
  if (idx < S0)           { nb = 0; outLen = 4096; j = idx; }
  else if (idx < S0 + S1) { nb = 1; outLen = 2048; j = idx - S0; }
  else                    { nb = 2; outLen = 1024; j = idx - S0 - S1; }

  const int d = (int)(j & 255);
  const int t = (int)((j >> 8) % outLen);
  const int b = (int)(j / ((long long)outLen * 256));

  float v;
  if (nb == 0) {
    v = y[(((size_t)b * LFULL + t) * NBR + 0) * D_DIM + d];
  } else {
    const float scale = (float)LFULL / (float)outLen;       // 2 or 4
    float src = ((float)t + 0.5f) * scale - 0.5f;
    src = fminf(fmaxf(src, 0.0f), (float)(LFULL - 1));
    int i0 = (int)src;
    int i1 = i0 + 1; if (i1 > LFULL - 1) i1 = LFULL - 1;
    float w = src - (float)i0;
    v = y[(((size_t)b * LFULL + i0) * NBR + nb) * D_DIM + d] * (1.0f - w)
      + y[(((size_t)b * LFULL + i1) * NBR + nb) * D_DIM + d] * w;
  }
  out[idx] = v;
}

// ---------------------------------------------------------------------------
extern "C" void kernel_launch(void* const* d_in, const int* in_sizes, int n_in,
                              void* d_out, int out_size, void* d_ws, size_t ws_size,
                              hipStream_t stream) {
  (void)in_sizes; (void)n_in; (void)out_size; (void)ws_size;

  const float* br0 = (const float*)d_in[0];
  const float* br1 = (const float*)d_in[1];
  const float* br2 = (const float*)d_in[2];
  const float* Wq  = (const float*)d_in[3];
  const float* bq  = (const float*)d_in[4];
  const float* Wk  = (const float*)d_in[5];
  const float* bk  = (const float*)d_in[6];
  const float* Wv  = (const float*)d_in[7];
  const float* bv  = (const float*)d_in[8];
  const float* Wo  = (const float*)d_in[9];
  const float* bo  = (const float*)d_in[10];

  unsigned short* Wt = (unsigned short*)d_ws;                  // 4*65536 bf16 = 512 KB
  float* y = (float*)((char*)d_ws + 4 * 65536 * sizeof(unsigned short));
  // y: 8*4096*3*256 f32 = ~100.7 MB

  prep_weights<<<1024, 256, 0, stream>>>(Wq, Wk, Wv, Wo, Wt);
  fused_attn<<<2048, 256, 0, stream>>>(br0, br1, br2, bq, bk, bv, bo, Wt, y);

  const long long total = 8LL * (4096 + 2048 + 1024) * 256;    // 14,680,064
  resample_out<<<(unsigned)(total / 256), 256, 0, stream>>>(y, (float*)d_out);
}
